// SparseMultiHeadAttention_14568529068646
// MI455X (gfx1250) — matrix operand order repaired and checked by reading, not compiled, not run on hardware
//
#include <hip/hip_runtime.h>
#include <hip/hip_bf16.h>

#define N_TOK   32768
#define CH      1024
#define NHEADS  16
#define HDIM    64

typedef __attribute__((ext_vector_type(16))) __bf16 v16bf;
typedef __attribute__((ext_vector_type(8)))  __bf16 bf16x8;
typedef __attribute__((ext_vector_type(8)))  float  v8f;

union U16 { v16bf v; bf16x8 h[2]; };
union BW  { unsigned int u; __bf16 b[2]; };

__device__ __forceinline__ v8f wmma_bf16(v16bf a, v16bf b, v8f c) {
    return __builtin_amdgcn_wmma_f32_16x16x32_bf16(false, a, false, b, (short)0, c, false, false);
}

// A-fragment (16x32, M x K), 16-bit: lanes 0-15 row l K{0..7,16..23}, lanes 16-31 row l K{8..15,24..31}
__device__ __forceinline__ v16bf load_frag_a(const __bf16* base, int row0, int k0, int stride, int lane) {
    int lm = lane & 15, lh = lane >> 4;
    const __bf16* p = base + (size_t)(row0 + lm) * stride + k0 + lh * 8;
    U16 u;
    u.h[0] = *(const bf16x8*)(p);
    u.h[1] = *(const bf16x8*)(p + 16);
    return u.v;
}

// B-fragment (32x16, K x N) from [col][k] array: lanes 0-15 col n K{0..7,16..23}, lanes 16-31 col n K{8..15,24..31}
__device__ __forceinline__ v16bf load_frag_b(const __bf16* base, int col0, int k0, int stride, int lane) {
    int lm = lane & 15, lh = lane >> 4;
    const __bf16* p = base + (size_t)(col0 + lm) * stride + k0 + lh * 8;
    U16 u;
    u.h[0] = *(const bf16x8*)(p);
    u.h[1] = *(const bf16x8*)(p + 16);
    return u.v;
}

// ---------------- weight convert + transpose:  dst[c][r] = (bf16) src[r][c] ----------------
__global__ void convert_transpose(const float* __restrict__ src, __bf16* __restrict__ dst,
                                  int rows, int cols) {
    long i = (long)blockIdx.x * 256 + threadIdx.x;
    long tot = (long)rows * cols;
    if (i >= tot) return;
    int c = (int)(i / rows);
    int r = (int)(i % rows);
    dst[i] = (__bf16)src[(size_t)r * cols + c];
}

// ---------------- cell map ----------------
__global__ void build_cellmap(const int* __restrict__ coords, int* __restrict__ cellmap) {
    int n = blockIdx.x * 256 + threadIdx.x;
    if (n >= N_TOK) return;
    int x = coords[n * 4 + 1], y = coords[n * 4 + 2], z = coords[n * 4 + 3];
    cellmap[(x * 32 + y) * 32 + z] = n;
}

// ---------------- WMMA GEMM:  C[M][Nc] = A[M][K] * Bt[Nc][K]^T + bias ----------------
#define LDS_STRIDE 40   // 32 + 8 pad (80B rows, 16B aligned)
template <bool A_IS_F32, bool OUT_BF16>
__global__ __launch_bounds__(256) void gemm_wmma(const void* __restrict__ A,
                                                 const __bf16* __restrict__ Bt,
                                                 const float* __restrict__ bias,
                                                 void* __restrict__ C,
                                                 int M, int Nc, int K) {
    __shared__ alignas(16) __bf16 As[128 * LDS_STRIDE];
    __shared__ alignas(16) __bf16 Bs[128 * LDS_STRIDE];

    int tid  = threadIdx.x;
    int lane = tid & 31;
    int wave = tid >> 5;
    int wm = wave >> 1;         // 0..3  (32 rows each)
    int wn = wave & 1;          // 0..1  (64 cols each)
    int rowbase = blockIdx.y * 128;
    int colbase = blockIdx.x * 128;

    int lrow = tid >> 1;        // 0..127
    int kp   = (tid & 1) * 16;  // 0 or 16

    v8f acc[2][4] = {};

    for (int k0 = 0; k0 < K; k0 += 32) {
        __syncthreads();
        // ---- load A tile [128][32] ----
        if (A_IS_F32) {
            const float* ap = (const float*)A + (size_t)(rowbase + lrow) * K + k0 + kp;
            float tmp[16];
            *(float4*)(tmp + 0)  = ((const float4*)ap)[0];
            *(float4*)(tmp + 4)  = ((const float4*)ap)[1];
            *(float4*)(tmp + 8)  = ((const float4*)ap)[2];
            *(float4*)(tmp + 12) = ((const float4*)ap)[3];
            bf16x8 v0, v1;
            #pragma unroll
            for (int j = 0; j < 8; ++j) { v0[j] = (__bf16)tmp[j]; v1[j] = (__bf16)tmp[8 + j]; }
            *(bf16x8*)&As[lrow * LDS_STRIDE + kp]     = v0;
            *(bf16x8*)&As[lrow * LDS_STRIDE + kp + 8] = v1;
        } else {
            const uint4* ap = (const uint4*)((const __bf16*)A + (size_t)(rowbase + lrow) * K + k0 + kp);
            *(uint4*)&As[lrow * LDS_STRIDE + kp] = ap[0];
            *(uint4*)&As[lrow * LDS_STRIDE + kp + 8] = ap[1];
        }
        // ---- load B tile [128 cols][32 k] from Bt[Nc][K] ----
        {
            const uint4* bp = (const uint4*)(Bt + (size_t)(colbase + lrow) * K + k0 + kp);
            *(uint4*)&Bs[lrow * LDS_STRIDE + kp] = bp[0];
            *(uint4*)&Bs[lrow * LDS_STRIDE + kp + 8] = bp[1];
        }
        __syncthreads();

        v16bf a0 = load_frag_a(As, wm * 32,      0, LDS_STRIDE, lane);
        v16bf a1 = load_frag_a(As, wm * 32 + 16, 0, LDS_STRIDE, lane);
        #pragma unroll
        for (int nt = 0; nt < 4; ++nt) {
            v16bf b = load_frag_b(Bs, wn * 64 + nt * 16, 0, LDS_STRIDE, lane);
            acc[0][nt] = wmma_bf16(a0, b, acc[0][nt]);
            acc[1][nt] = wmma_bf16(a1, b, acc[1][nt]);
        }
    }

    int lm = lane & 15, lh = lane >> 4;
    #pragma unroll
    for (int mt = 0; mt < 2; ++mt)
        #pragma unroll
        for (int nt = 0; nt < 4; ++nt) {
            int col = colbase + wn * 64 + nt * 16 + lm;
            float bv = bias[col];
            #pragma unroll
            for (int r = 0; r < 8; ++r) {
                int rowg = rowbase + wm * 32 + mt * 16 + r + 8 * lh;
                float val = acc[mt][nt][r] + bv;
                if (OUT_BF16) ((__bf16*)C)[(size_t)rowg * Nc + col] = (__bf16)val;
                else          ((float*)C)[(size_t)rowg * Nc + col] = val;
            }
        }
}

// ---------------- RMS norm q,k in place (wave per token-head) ----------------
__global__ __launch_bounds__(256) void rmsnorm_qk(__bf16* __restrict__ qkv,
                                                  const float* __restrict__ gq,
                                                  const float* __restrict__ gk) {
    int wid  = blockIdx.x * 8 + (threadIdx.x >> 5);
    int lane = threadIdx.x & 31;
    int tok  = wid >> 4;
    int head = wid & 15;
    if (tok >= N_TOK) return;
    #pragma unroll
    for (int which = 0; which < 2; ++which) {
        __bf16* base = qkv + (size_t)tok * 3072 + which * 1024 + head * 64;
        const float* g = (which ? gk : gq) + head * 64;
        BW u; u.u = ((unsigned int*)base)[lane];
        float f0 = (float)u.b[0], f1 = (float)u.b[1];
        float ss = f0 * f0 + f1 * f1;
        for (int d = 1; d < 32; d <<= 1) ss += __shfl_xor(ss, d, 32);
        float sc = 8.0f / fmaxf(sqrtf(ss), 1e-12f);
        u.b[0] = (__bf16)(f0 * sc * g[2 * lane]);
        u.b[1] = (__bf16)(f1 * sc * g[2 * lane + 1]);
        ((unsigned int*)base)[lane] = u.u;
    }
}

// ---------------- windowed attention: 1 wave per (window, head) ----------------
#define APAD 72   // row pad (144B): conflict-free 16-row b128 fragment reads
__global__ __launch_bounds__(32) void win_attn(const __bf16* __restrict__ qkv,
                                               const int* __restrict__ cellmap,
                                               __bf16* __restrict__ hout) {
    __shared__ alignas(16) __bf16 Qs[64 * APAD];
    __shared__ alignas(16) __bf16 Ks[64 * APAD];
    __shared__ alignas(16) __bf16 Vt[64 * APAD];
    __shared__ alignas(16) __bf16 Ps[16 * APAD];
    __shared__ int toks[64];

    int lane = threadIdx.x;
    int b = blockIdx.x;
    int head, outslot, x0, y0, z0, nx, ny, nz;
    if (b < 4096) {                        // shift (0,0,0): heads 8..15 -> out slots 0..7
        int w = b >> 3, h8 = b & 7;
        head = 8 + h8; outslot = h8;
        int wz = w & 7, wy = (w >> 3) & 7, wx = w >> 6;
        x0 = wx * 4; y0 = wy * 4; z0 = wz * 4; nx = ny = nz = 4;
    } else {                               // shift (2,2,2): heads 0..7 -> out slots 8..15
        int bb = b - 4096;
        int w = bb >> 3, h8 = bb & 7;
        head = h8; outslot = 8 + h8;
        int wz = w % 9, wy = (w / 9) % 9, wx = w / 81;
        auto rng = [](int wv, int& o, int& n) {
            int lo = wv * 4 - 2; if (lo < 0) lo = 0;
            int hi = wv * 4 + 2; if (hi > 32) hi = 32;
            o = lo; n = hi - lo;
        };
        rng(wx, x0, nx); rng(wy, y0, ny); rng(wz, z0, nz);
    }
    int m = nx * ny * nz;

    for (int i = lane; i < 64 * APAD; i += 32) {
        Qs[i] = (__bf16)0.f; Ks[i] = (__bf16)0.f; Vt[i] = (__bf16)0.f;
    }
    for (int i = lane; i < m; i += 32) {
        int iz = i % nz, t = i / nz;
        int iy = t % ny, ix = t / ny;
        toks[i] = cellmap[((x0 + ix) * 32 + (y0 + iy)) * 32 + (z0 + iz)];
    }
    __syncthreads();

    for (int r = 0; r < m; ++r) {
        size_t base = (size_t)toks[r] * 3072 + head * 64;
        unsigned int qw = ((const unsigned int*)(qkv + base))[lane];
        unsigned int kw = ((const unsigned int*)(qkv + base + 1024))[lane];
        unsigned int vw = ((const unsigned int*)(qkv + base + 2048))[lane];
        *(unsigned int*)&Qs[r * APAD + lane * 2] = qw;
        *(unsigned int*)&Ks[r * APAD + lane * 2] = kw;
        BW u; u.u = vw;
        Vt[(2 * lane) * APAD + r]     = u.b[0];
        Vt[(2 * lane + 1) * APAD + r] = u.b[1];
    }
    __syncthreads();

    int lm = lane & 15, lh = lane >> 4;

    for (int qt = 0; qt < 4; ++qt) {
        v16bf a0 = load_frag_a(Qs, qt * 16, 0, APAD, lane);
        v16bf a1 = load_frag_a(Qs, qt * 16, 32, APAD, lane);
        v8f S[4];
        #pragma unroll
        for (int kt = 0; kt < 4; ++kt) {
            v8f acc = {};
            acc = wmma_bf16(a0, load_frag_b(Ks, kt * 16, 0, APAD, lane), acc);
            acc = wmma_bf16(a1, load_frag_b(Ks, kt * 16, 32, APAD, lane), acc);
            S[kt] = acc;
        }
        // scale + validity mask (matches reference -1e9 masking)
        #pragma unroll
        for (int kt = 0; kt < 4; ++kt) {
            bool valid = (kt * 16 + lm) < m;
            #pragma unroll
            for (int r = 0; r < 8; ++r) {
                float s = S[kt][r] * 0.125f;
                S[kt][r] = valid ? s : -1e9f;
            }
        }
        // row-wise softmax (rows live across lanes 0-15 / 16-31 per accumulator slot)
        #pragma unroll
        for (int r = 0; r < 8; ++r) {
            float mx = fmaxf(fmaxf(S[0][r], S[1][r]), fmaxf(S[2][r], S[3][r]));
            for (int d = 1; d < 16; d <<= 1) mx = fmaxf(mx, __shfl_xor(mx, d, 32));
            float e[4];
            #pragma unroll
            for (int kt = 0; kt < 4; ++kt) e[kt] = __expf(S[kt][r] - mx);
            float sm = e[0] + e[1] + e[2] + e[3];
            for (int d = 1; d < 16; d <<= 1) sm += __shfl_xor(sm, d, 32);
            float inv = 1.0f / sm;
            #pragma unroll
            for (int kt = 0; kt < 4; ++kt) S[kt][r] = e[kt] * inv;
        }
        __syncthreads();
        #pragma unroll
        for (int kt = 0; kt < 4; ++kt)
            #pragma unroll
            for (int r = 0; r < 8; ++r)
                Ps[(r + 8 * lh) * APAD + kt * 16 + lm] = (__bf16)S[kt][r];
        __syncthreads();

        v8f O[4] = {};
        #pragma unroll
        for (int ks = 0; ks < 2; ++ks) {
            v16bf pa = load_frag_a(Ps, 0, ks * 32, APAD, lane);
            #pragma unroll
            for (int dt = 0; dt < 4; ++dt)
                O[dt] = wmma_bf16(pa, load_frag_b(Vt, dt * 16, ks * 32, APAD, lane), O[dt]);
        }
        #pragma unroll
        for (int dt = 0; dt < 4; ++dt)
            #pragma unroll
            for (int r = 0; r < 8; ++r) {
                int qi = qt * 16 + r + 8 * lh;
                if (qi < m) {
                    size_t o = (size_t)toks[qi] * 1024 + outslot * 64 + dt * 16 + lm;
                    hout[o] = (__bf16)O[dt][r];
                }
            }
        __syncthreads();
    }
}

// ---------------- host launcher ----------------
extern "C" void kernel_launch(void* const* d_in, const int* in_sizes, int n_in,
                              void* d_out, int out_size, void* d_ws, size_t ws_size,
                              hipStream_t stream) {
    const float* x       = (const float*)d_in[0];
    const int*   coords  = (const int*)d_in[1];
    const float* W_qkv   = (const float*)d_in[2];
    const float* b_qkv   = (const float*)d_in[3];
    const float* gamma_q = (const float*)d_in[4];
    const float* gamma_k = (const float*)d_in[5];
    const float* W_out   = (const float*)d_in[6];
    const float* b_out   = (const float*)d_in[7];
    float* out = (float*)d_out;

    size_t off = 0;
    auto alloc = [&](size_t bytes) -> void* {
        void* p = (char*)d_ws + off;
        off += (bytes + 255) & ~(size_t)255;
        return p;
    };
    __bf16* Wqkv_t = (__bf16*)alloc((size_t)3072 * 1024 * 2);
    __bf16* Wout_t = (__bf16*)alloc((size_t)1024 * 1024 * 2);
    __bf16* qkv_bf = (__bf16*)alloc((size_t)N_TOK * 3072 * 2);
    __bf16* h_bf   = (__bf16*)alloc((size_t)N_TOK * 1024 * 2);
    int*    cellmap = (int*)alloc((size_t)N_TOK * 4);

    convert_transpose<<<(3072 * 1024 + 255) / 256, 256, 0, stream>>>(W_qkv, Wqkv_t, 1024, 3072);
    convert_transpose<<<(1024 * 1024 + 255) / 256, 256, 0, stream>>>(W_out, Wout_t, 1024, 1024);
    build_cellmap<<<N_TOK / 256, 256, 0, stream>>>(coords, cellmap);

    gemm_wmma<true, true><<<dim3(3072 / 128, N_TOK / 128), 256, 0, stream>>>(
        (const void*)x, Wqkv_t, b_qkv, (void*)qkv_bf, N_TOK, 3072, 1024);

    rmsnorm_qk<<<(N_TOK * 16) / 8, 256, 0, stream>>>(qkv_bf, gamma_q, gamma_k);

    win_attn<<<4096 + 5832, 32, 0, stream>>>(qkv_bf, cellmap, h_bf);

    gemm_wmma<false, false><<<dim3(1024 / 128, N_TOK / 128), 256, 0, stream>>>(
        (const void*)h_bf, Wout_t, b_out, (void*)out, N_TOK, 1024, 1024);
}
